// CausalSelfAttention_87385404604866
// MI455X (gfx1250) — matrix operand order repaired and checked by reading, not compiled, not run on hardware
//
#include <hip/hip_runtime.h>
#include <hip/hip_bf16.h>

typedef _Float16 v16h __attribute__((ext_vector_type(16)));
typedef _Float16 v8h  __attribute__((ext_vector_type(8)));
typedef float    v8f  __attribute__((ext_vector_type(8)));
typedef int      v4i_ __attribute__((ext_vector_type(4)));

#define TILE_M 128
#define TILE_N 128
#define TILE_K 32
#define LDS_STRIDE 40   // 32 data + 8 pad halves; row stride 80B (16B aligned)

// ---- CDNA5 async global->LDS copy (ASYNCcnt), with graceful fallback ----
#if __has_builtin(__builtin_amdgcn_global_load_async_to_lds_b128)
#define HAVE_ASYNC_LDS 1
#else
#define HAVE_ASYNC_LDS 0
#endif

#if HAVE_ASYNC_LDS
__device__ __forceinline__ void async_cp16(const void* g, void* l) {
    // copies 16 bytes per lane, tracked by ASYNCcnt, no VGPR data round-trip
    __builtin_amdgcn_global_load_async_to_lds_b128(
        (__attribute__((address_space(1))) v4i_*)(void*)g,
        (__attribute__((address_space(3))) v4i_*)l,
        /*imm offset*/0, /*cpol*/0);
}
__device__ __forceinline__ void wait_async0() {
#if __has_builtin(__builtin_amdgcn_s_wait_asynccnt)
    __builtin_amdgcn_s_wait_asynccnt(0);
#else
    asm volatile("s_wait_asynccnt 0x0" ::: "memory");
#endif
}
#else
__device__ __forceinline__ void async_cp16(const void*, void*) {}
__device__ __forceinline__ void wait_async0() {}
#endif

// C[M,N] = A[M,K] * (BT ? B[N,K]^T : B[K,N]) (+ bias), f16 WMMA, f32 accum.
// Double-buffered LDS. f16 operands are prefetched with async global->LDS
// copies; f32 operands go through a software-pipelined load/cvt/ds_store path.
// 256 threads = 8 waves; wave (wm,wn) owns 32x64 = 2x4 WMMA 16x16 tiles.
// All dims are multiples of the tile sizes for every call here.
template<typename TA, typename TB, typename TC, bool BT, bool HAS_BIAS>
__global__ __launch_bounds__(256)
void gemm_wmma(const TA* __restrict__ A, int lda,
               const TB* __restrict__ B, int ldb,
               const float* __restrict__ bias,
               TC* __restrict__ C, int ldc,
               int K, int causal)
{
    constexpr bool A_ASYNC = HAVE_ASYNC_LDS && (sizeof(TA) == 2);
    constexpr bool B_ASYNC = HAVE_ASYNC_LDS && BT && (sizeof(TB) == 2);

    __shared__ _Float16 Alds[2][TILE_M * LDS_STRIDE];
    __shared__ _Float16 Blds[2][TILE_N * LDS_STRIDE];

    const int rowBase = blockIdx.y * TILE_M;
    const int colBase = blockIdx.x * TILE_N;

    // fully-masked causal tile: skip (softmax pass writes the zeros)
    if (causal && colBase > rowBase + (TILE_M - 1)) return;

    const int tid   = threadIdx.x;
    const int lane  = tid & 31;
    const int wave  = tid >> 5;      // 0..7
    const int wm    = wave >> 1;     // 0..3 : 32-row strip
    const int wn    = wave & 1;      // 0..1 : 64-col strip
    const int lhalf = lane >> 4;     // which half-wave
    const int l15   = lane & 15;

    v8f acc[2][4];
    #pragma unroll
    for (int i = 0; i < 2; ++i)
        #pragma unroll
        for (int j = 0; j < 4; ++j)
            #pragma unroll
            for (int r = 0; r < 8; ++r)
                acc[i][j][r] = 0.0f;

    // tile loaders: A (and B when BT): 2 threads/row * 16 contiguous elems
    const int arow = tid >> 1;
    const int acol = (tid & 1) * 16;
    // B when !BT: 8 threads per k-row * 16 elems, transposed into LDS
    const int bkrow = tid >> 3;
    const int bnb   = (tid & 7) * 16;

    float ar[16], br[16];   // staging regs for sync (f32->f16) paths

    auto prefetchA = [&](int k0, int buf) {
        if constexpr (A_ASYNC) {
            const TA* src = A + (size_t)(rowBase + arow) * lda + k0 + acol;
            _Float16* dst = &Alds[buf][arow * LDS_STRIDE + acol];
            async_cp16(src, dst);
            async_cp16(src + 8, dst + 8);
        }
    };
    auto prefetchB = [&](int k0, int buf) {
        if constexpr (B_ASYNC) {
            const TB* src = B + (size_t)(colBase + arow) * ldb + k0 + acol;
            _Float16* dst = &Blds[buf][arow * LDS_STRIDE + acol];
            async_cp16(src, dst);
            async_cp16(src + 8, dst + 8);
        }
    };
    auto loadA_sync = [&](int k0) {
        if constexpr (!A_ASYNC) {
            const TA* src = A + (size_t)(rowBase + arow) * lda + k0 + acol;
            #pragma unroll
            for (int j = 0; j < 16; ++j) ar[j] = (float)src[j];
        }
    };
    auto loadB_sync = [&](int k0) {
        if constexpr (!B_ASYNC) {
            const TB* src = BT
                ? (B + (size_t)(colBase + arow) * ldb + k0 + acol)
                : (B + (size_t)(k0 + bkrow) * ldb + colBase + bnb);
            #pragma unroll
            for (int j = 0; j < 16; ++j) br[j] = (float)src[j];
        }
    };
    auto storeA_sync = [&](int buf) {
        if constexpr (!A_ASYNC) {
            _Float16* dst = &Alds[buf][arow * LDS_STRIDE + acol];
            #pragma unroll
            for (int j = 0; j < 16; ++j) dst[j] = (_Float16)ar[j];
        }
    };
    auto storeB_sync = [&](int buf) {
        if constexpr (!B_ASYNC) {
            if constexpr (BT) {
                _Float16* dst = &Blds[buf][arow * LDS_STRIDE + acol];
                #pragma unroll
                for (int j = 0; j < 16; ++j) dst[j] = (_Float16)br[j];
            } else {
                #pragma unroll
                for (int j = 0; j < 16; ++j)
                    Blds[buf][(bnb + j) * LDS_STRIDE + bkrow] = (_Float16)br[j];
            }
        }
    };
    auto compute = [&](int buf) {
        // A frag (ISA 7.12.2, 16-bit A 16x32): lane l15 = M row,
        // halves 0..7 -> K = 8*lhalf + h, halves 8..15 -> K = 16 + 8*lhalf + h
        v16h afrag[2];
        #pragma unroll
        for (int i = 0; i < 2; ++i) {
            const _Float16* ap = &Alds[buf][(wm * 32 + i * 16 + l15) * LDS_STRIDE];
            const int kb0 = lhalf * 8;
            #pragma unroll
            for (int h = 0; h < 8; ++h) {
                afrag[i][h]     = ap[kb0 + h];
                afrag[i][h + 8] = ap[16 + kb0 + h];
            }
        }
        // B frag: lane l15 = N col, halves 0..7 -> K = 8*lhalf + h, halves 8..15 -> K = 16 + 8*lhalf + h
        #pragma unroll
        for (int j = 0; j < 4; ++j) {
            const _Float16* bp =
                &Blds[buf][(wn * 64 + j * 16 + l15) * LDS_STRIDE + lhalf * 8];
            v16h bfrag;
            #pragma unroll
            for (int h = 0; h < 8; ++h) { bfrag[h] = bp[h]; bfrag[h + 8] = bp[16 + h]; }
            #pragma unroll
            for (int i = 0; i < 2; ++i) {
                acc[i][j] = __builtin_amdgcn_wmma_f32_16x16x32_f16(
                    false, afrag[i], false, bfrag,
                    (short)0, acc[i][j], false, false);
            }
        }
    };

    const int nIter = K / TILE_K;

    // prologue: stage chunk 0 into buffer 0
    prefetchA(0, 0);
    prefetchB(0, 0);
    loadA_sync(0);
    loadB_sync(0);
    storeA_sync(0);
    storeB_sync(0);

    for (int it = 0; it < nIter; ++it) {
        const int buf = it & 1, nbuf = buf ^ 1;
        if constexpr (A_ASYNC || B_ASYNC) wait_async0();
        __syncthreads();                 // buf ready for all waves
        const int k1 = (it + 1) * TILE_K;
        if (it + 1 < nIter) {            // prefetch next chunk into other buffer
            prefetchA(k1, nbuf);
            prefetchB(k1, nbuf);
            loadA_sync(k1);              // issue global loads early (no stall yet)
            loadB_sync(k1);
        }
        compute(buf);                    // 8x v_wmma from LDS
        if (it + 1 < nIter) {            // cvt + ds_store after compute
            storeA_sync(nbuf);
            storeB_sync(nbuf);
        }
    }

    // C/D layout: vgpr r, lane -> m = r + 8*lhalf, n = l15
    #pragma unroll
    for (int i = 0; i < 2; ++i) {
        #pragma unroll
        for (int j = 0; j < 4; ++j) {
            const int gn = colBase + wn * 64 + j * 16 + l15;
            float bv = 0.0f;
            if constexpr (HAS_BIAS) bv = bias[gn];
            #pragma unroll
            for (int r = 0; r < 8; ++r) {
                const int gm = rowBase + wm * 32 + i * 16 + r + lhalf * 8;
                C[(size_t)gm * ldc + gn] = (TC)(acc[i][j][r] + bv);
            }
        }
    }
}

// Single-pass in-place causal softmax over row blockIdx.x of S [T x T] f16.
// T = 4096, 256 threads: each thread owns exactly 16 elements (2 x v8h),
// kept in registers: one read, one write, exp computed once.
__global__ __launch_bounds__(256)
void causal_softmax_inplace(_Float16* __restrict__ S, int T)
{
    const int row = blockIdx.x;
    const int L   = row + 1;            // valid columns [0, L)
    const int tid = threadIdx.x;
    __shared__ float red[256];
    v8h* rowv = (v8h*)(S + (size_t)row * T);   // 512 chunks of 8

    v8h c[2];
    c[0] = rowv[tid];
    c[1] = rowv[tid + 256];

    float vals[16];
    float mx = -3.0e38f;
    #pragma unroll
    for (int q = 0; q < 2; ++q) {
        const int jb = (tid + q * 256) * 8;
        #pragma unroll
        for (int e = 0; e < 8; ++e) {
            const float v = (jb + e < L) ? (float)c[q][e] : -3.0e38f;
            vals[q * 8 + e] = v;
            mx = fmaxf(mx, v);
        }
    }
    red[tid] = mx;
    __syncthreads();
    for (int s = 128; s > 0; s >>= 1) {
        if (tid < s) red[tid] = fmaxf(red[tid], red[tid + s]);
        __syncthreads();
    }
    mx = red[0];
    __syncthreads();

    float sum = 0.0f;
    #pragma unroll
    for (int e = 0; e < 16; ++e) {
        const float ev = (vals[e] > -1.0e38f) ? __expf(vals[e] - mx) : 0.0f;
        vals[e] = ev;
        sum += ev;
    }
    red[tid] = sum;
    __syncthreads();
    for (int s = 128; s > 0; s >>= 1) {
        if (tid < s) red[tid] += red[tid + s];
        __syncthreads();
    }
    const float inv = 1.0f / red[0];

    #pragma unroll
    for (int q = 0; q < 2; ++q) {
        v8h o;
        #pragma unroll
        for (int e = 0; e < 8; ++e) o[e] = (_Float16)(vals[q * 8 + e] * inv);
        rowv[tid + q * 256] = o;
    }
}

extern "C" void kernel_launch(void* const* d_in, const int* in_sizes, int n_in,
                              void* d_out, int out_size, void* d_ws, size_t ws_size,
                              hipStream_t stream) {
    const float* x     = (const float*)d_in[0];   // [T, D]
    const float* Wqkv  = (const float*)d_in[1];   // [3D, D]
    const float* bqkv  = (const float*)d_in[2];   // [3D]
    const float* Wproj = (const float*)d_in[3];   // [D, D]
    const float* bproj = (const float*)d_in[4];   // [D]
    float* out = (float*)d_out;                   // [T, D] f32

    const int T = 4096, D = 2048;

    char* ws = (char*)d_ws;
    _Float16* qkvh = (_Float16*)ws;                                   // [T, 3D] 48MB
    _Float16* S    = (_Float16*)(ws + (size_t)T * 3 * D * 2);         // [T, T]  32MB
    _Float16* attn = (_Float16*)(ws + (size_t)T * 3 * D * 2
                                    + (size_t)T * T * 2);             // [T, D]  16MB

    dim3 blk(256);

    // 1) qkv = x @ Wqkv^T + bqkv      (f32 in -> f16 out; sync cvt path)
    gemm_wmma<float, float, _Float16, true, true>
        <<<dim3(3 * D / TILE_N, T / TILE_M), blk, 0, stream>>>(
            x, D, Wqkv, D, bqkv, qkvh, 3 * D, D, 0);

    // 2) S = k @ q^T   (causal tiles only; async LDS prefetch both operands)
    gemm_wmma<_Float16, _Float16, _Float16, true, false>
        <<<dim3(T / TILE_N, T / TILE_M), blk, 0, stream>>>(
            qkvh + D, 3 * D, qkvh, 3 * D, nullptr, S, T, D, 1);

    // 3) causal softmax in place -> P (f16), zeros above diagonal
    causal_softmax_inplace<<<dim3(T), blk, 0, stream>>>(S, T);

    // 4) attn = P @ v   (A async; B [K,N] transposed into LDS sync)
    gemm_wmma<_Float16, _Float16, _Float16, false, false>
        <<<dim3(D / TILE_N, T / TILE_M), blk, 0, stream>>>(
            S, T, qkvh + 2 * D, 3 * D, nullptr, attn, D, T, 0);

    // 5) out = attn @ Wproj^T + bproj (A async; B f32 sync; f32 out)
    gemm_wmma<_Float16, float, float, true, true>
        <<<dim3(D / TILE_N, T / TILE_M), blk, 0, stream>>>(
            attn, D, Wproj, D, bproj, out, D, D, 0);
}
